// EfficientLinearAttention_65755949302268
// MI455X (gfx1250) — matrix operand order repaired and checked by reading, not compiled, not run on hardware
//
#include <hip/hip_runtime.h>

// ---------------------------------------------------------------------------
// EfficientLinearAttention for MI455X (gfx1250), bf16 WMMA pipeline, v2:
// double-buffered async-to-LDS staging (ASYNCcnt) feeding v_wmma_f32_16x16x32_bf16.
// B=16, N=4096, C=768, H=12, D=64.
// ---------------------------------------------------------------------------

typedef __bf16 bf16_t;
typedef __attribute__((ext_vector_type(16))) __bf16 v16bf;
typedef __attribute__((ext_vector_type(8)))  __bf16 v8bf;
typedef __attribute__((ext_vector_type(4)))  __bf16 v4bf;
typedef __attribute__((ext_vector_type(8)))  float  v8f;
typedef __attribute__((ext_vector_type(4)))  float  v4f;

#define KB   16
#define KH   12
#define KN   4096
#define KC   768
#define KD   64
#define KM   (KB * KN)      // 65536 rows of x
#define KCQ  (3 * KC)       // 2304 qkv outputs

// ---- workspace layout (bytes) ---------------------------------------------
static constexpr size_t SZ_XBF   = (size_t)KM * KC * 2;           // 96 MB
static constexpr size_t SZ_WQKV  = (size_t)KCQ * KC * 2;
static constexpr size_t SZ_WPROJ = (size_t)KC * KC * 2;
static constexpr size_t SZ_QWS   = (size_t)KB * KH * KN * KD * 2; // 96 MB
static constexpr size_t SZ_KT    = SZ_QWS;
static constexpr size_t SZ_VT    = SZ_QWS;
static constexpr size_t SZ_KV    = (size_t)KB * KH * KD * KD * 2;
static constexpr size_t SZ_KSUM  = (size_t)KB * KH * KD * 4;

static constexpr size_t OFF_XBF   = 0;
static constexpr size_t OFF_WQKV  = OFF_XBF + SZ_XBF;
static constexpr size_t OFF_WPROJ = OFF_WQKV + SZ_WQKV;
static constexpr size_t OFF_QWS   = OFF_WPROJ + SZ_WPROJ;
static constexpr size_t OFF_KT    = OFF_QWS + SZ_QWS;
static constexpr size_t OFF_VT    = OFF_KT + SZ_KT;
static constexpr size_t OFF_KV    = OFF_VT + SZ_VT;
static constexpr size_t OFF_KSUM  = OFF_KV + SZ_KV;
static constexpr size_t OFF_ATTN  = OFF_XBF;   // alias: xbf dead after qkv GEMM

// ---- async copy (gfx1250): 16B global -> LDS, tracked with ASYNCcnt -------
__device__ __forceinline__ void async_b128(unsigned lds_off, const bf16_t* g) {
  asm volatile("global_load_async_to_lds_b128 %0, %1, off"
               :: "v"(lds_off), "v"((unsigned long long)(size_t)g)
               : "memory");
}
__device__ __forceinline__ void wait_async0() {
  asm volatile("s_wait_asynccnt 0x0" ::: "memory");
}
__device__ __forceinline__ unsigned lds_off32(const bf16_t* p) {
  return (unsigned)(size_t)p;   // LDS generic addr: offset lives in [31:0]
}

// ---- WMMA operand loaders (match gfx1250 16-bit A / B VGPR striping) ------
// A (16x32, row-major, ld elements): lane l -> row l&15, K base (l>>4)*8 and +16.
__device__ __forceinline__ v16bf load_a(const bf16_t* base, int ld, int lane) {
  const int row = lane & 15;
  const int kb  = (lane >> 4) << 3;
  const bf16_t* p = base + (size_t)row * ld + kb;
  v8bf lo = *(const v8bf*)p;
  v8bf hi = *(const v8bf*)(p + 16);
  v16bf r;
#pragma unroll
  for (int i = 0; i < 8; ++i) { r[i] = lo[i]; r[i + 8] = hi[i]; }
  return r;
}

// B (32x16) from B^T memory (rows = output column):
// lane l -> column l&15, K base (l>>4)*8 and +16.  Two 16-byte loads.
__device__ __forceinline__ v16bf load_bT(const bf16_t* base, int ld, int lane) {
  const int col = lane & 15;
  const int kb  = (lane >> 4) << 3;
  const bf16_t* p = base + (size_t)col * ld + kb;
  v8bf lo = *(const v8bf*)p;
  v8bf hi = *(const v8bf*)(p + 16);
  v16bf r;
#pragma unroll
  for (int i = 0; i < 8; ++i) { r[i] = lo[i]; r[i + 8] = hi[i]; }
  return r;
}

__device__ __forceinline__ v8f wmma_bf16(v16bf a, v16bf b, v8f c) {
  return __builtin_amdgcn_wmma_f32_16x16x32_bf16(false, a, false, b, (short)0, c,
                                                 false, false);
}

// ---------------------------------------------------------------------------
// fp32 -> bf16 conversion (vector x4)
// ---------------------------------------------------------------------------
__global__ __launch_bounds__(256)
void f32_to_bf16_kernel(const float* __restrict__ in, bf16_t* __restrict__ out, int n) {
  int i = (blockIdx.x * blockDim.x + threadIdx.x) * 4;
  if (i < n) {
    v4f f = *(const v4f*)(in + i);
    v4bf o;
#pragma unroll
    for (int j = 0; j < 4; ++j) o[j] = (bf16_t)f[j];
    *(v4bf*)(out + i) = o;
  }
}

// ---------------------------------------------------------------------------
// Shared GEMM inner pipeline: block 256 thr (8 waves, 2x4), tile 64(M)x128(N),
// BK=32, double-buffered async LDS staging.  Wave tile 32x32 (4 accumulators).
//   lsA: [2][64*32] bf16 (4 KB/buf), lsB: [2][128*32] bf16 (8 KB/buf) -> 24 KB.
// ---------------------------------------------------------------------------
__device__ __forceinline__ void issue_tileA(const bf16_t* A, int ld, int m0, int k,
                                            bf16_t* lsA, int tid) {
  const int r = tid >> 2, c = (tid & 3) * 8;           // 64 rows x 32 cols, 16B each
  async_b128(lds_off32(lsA + r * 32 + c), A + (size_t)(m0 + r) * ld + k + c);
}
__device__ __forceinline__ void issue_tileB(const bf16_t* W, int ld, int n0, int k,
                                            bf16_t* lsB, int tid) {
  const int r = tid >> 1, c = (tid & 1) * 16;          // 128 rows x 32 cols, 2x16B
  const bf16_t* g = W + (size_t)(n0 + r) * ld + k + c;
  unsigned l = lds_off32(lsB + r * 32 + c);
  async_b128(l, g);
  async_b128(l + 16, g + 8);
}

__device__ __forceinline__ void gemm_pipeline(const bf16_t* A, const bf16_t* W, int K,
                                              bf16_t* lsA, bf16_t* lsB,
                                              int m0, int n0, int tid, int lane,
                                              int wm, int wn, v8f acc[2][2]) {
  issue_tileA(A, K, m0, 0, lsA, tid);
  issue_tileB(W, K, n0, 0, lsB, tid);
  const int nk = K / 32;
  for (int kc = 0; kc < nk; ++kc) {
    const int cur = kc & 1;
    wait_async0();        // this thread's copies for chunk kc have landed
    __syncthreads();      // everyone's copies landed; prev buffer reads all done
    if (kc + 1 < nk) {    // overlap: DMA chunk kc+1 while computing chunk kc
      issue_tileA(A, K, m0, (kc + 1) * 32, lsA + (1 - cur) * (64 * 32), tid);
      issue_tileB(W, K, n0, (kc + 1) * 32, lsB + (1 - cur) * (128 * 32), tid);
    }
    const bf16_t* pA = lsA + cur * (64 * 32) + (wm * 32) * 32;
    const bf16_t* pB = lsB + cur * (128 * 32) + (wn * 32) * 32;
    v16bf a0 = load_a(pA, 32, lane);
    v16bf a1 = load_a(pA + 16 * 32, 32, lane);
    v16bf b0 = load_bT(pB, 32, lane);
    v16bf b1 = load_bT(pB + 16 * 32, 32, lane);
    acc[0][0] = wmma_bf16(a0, b0, acc[0][0]);
    acc[0][1] = wmma_bf16(a0, b1, acc[0][1]);
    acc[1][0] = wmma_bf16(a1, b0, acc[1][0]);
    acc[1][1] = wmma_bf16(a1, b1, acc[1][1]);
  }
}

// ---------------------------------------------------------------------------
// QKV GEMM: [65536 x 768] @ [2304 x 768]^T, fused ReLU(q/t), ReLU(k/t),
// transposed k/v stores.
// ---------------------------------------------------------------------------
__global__ __launch_bounds__(256)
void qkv_gemm_kernel(const bf16_t* __restrict__ X,    // [M, C]
                     const bf16_t* __restrict__ W,    // [3C, C]  (B^T layout)
                     const float*  __restrict__ temperature, // [H]
                     bf16_t* __restrict__ q_ws,       // [B,H,N,D]
                     bf16_t* __restrict__ kT_ws,      // [B,H,D,N]
                     bf16_t* __restrict__ vT_ws) {    // [B,H,D,N]
  __shared__ bf16_t lsA[2][64 * 32];
  __shared__ bf16_t lsB[2][128 * 32];
  const int tid  = threadIdx.x;
  const int lane = tid & 31;
  const int wave = tid >> 5;
  const int wm   = wave >> 2, wn = wave & 3;
  const int m0   = blockIdx.y * 64;
  const int n0   = blockIdx.x * 128;

  v8f acc[2][2] = {};
  gemm_pipeline(X, W, KC, &lsA[0][0], &lsB[0][0], m0, n0, tid, lane, wm, wn, acc);

#pragma unroll
  for (int mi = 0; mi < 2; ++mi)
#pragma unroll
    for (int ni = 0; ni < 2; ++ni) {
      v8f a = acc[mi][ni];
      const int col = n0 + wn * 32 + ni * 16 + (lane & 15);
      const int sec = col / KC;                  // 0=q 1=k 2=v (uniform per tile)
      const int cc  = col - sec * KC;
      const int h   = cc >> 6;
      const int d   = cc & 63;
      const float tinv = 1.0f / temperature[h];
#pragma unroll
      for (int i = 0; i < 8; ++i) {
        const int m = m0 + wm * 32 + mi * 16 + i + ((lane >> 4) << 3);
        const int b = m >> 12;
        const int n = m & 4095;
        const size_t bh = (size_t)(b * KH + h);
        float val = a[i];
        if (sec == 0) {
          val *= tinv; val = val > 0.f ? val : 0.f;
          q_ws[(bh * KN + n) * KD + d] = (bf16_t)val;
        } else if (sec == 1) {
          val *= tinv; val = val > 0.f ? val : 0.f;
          kT_ws[(bh * KD + d) * KN + n] = (bf16_t)val;
        } else {
          vT_ws[(bh * KD + d) * KN + n] = (bf16_t)val;
        }
      }
    }
}

// ---------------------------------------------------------------------------
// kv[d,e] = sum_n k[n,d] v[n,e] per (b,h).  Block (16 waves) per (b,h), async
// double-buffered k/v chunk staging (64x32 each), wave owns one 16x16 tile.
// Output transposed: kvT[b,h,e,d].
// ---------------------------------------------------------------------------
__global__ __launch_bounds__(512)
void kv_kernel(const bf16_t* __restrict__ kT,    // [BH, D, N]
               const bf16_t* __restrict__ vT,    // [BH, D, N]
               bf16_t* __restrict__ kvT) {       // [BH, E, D]
  __shared__ bf16_t lsK[2][64 * 32];
  __shared__ bf16_t lsV[2][64 * 32];
  const int bh   = blockIdx.x;
  const int tid  = threadIdx.x;
  const int lane = tid & 31;
  const int wave = tid >> 5;
  const int d0   = (wave >> 2) * 16;
  const int e0   = (wave & 3) * 16;

  const bf16_t* kbase = kT + (size_t)bh * KD * KN;   // rows d, ld = N
  const bf16_t* vbase = vT + (size_t)bh * KD * KN;   // rows e, ld = N

  // thread -> one 16B async transfer per chunk (512 thr cover 2 x 4 KB)
  const int half = tid >> 8;                 // 0: K tile, 1: V tile
  const int t8   = tid & 255;
  const int cr   = t8 >> 2, ccol = (t8 & 3) * 8;
  const bf16_t* gsrc = (half ? vbase : kbase) + (size_t)cr * KN + ccol;
  bf16_t* ldst0 = (half ? &lsV[0][0] : &lsK[0][0]) + cr * 32 + ccol;
  bf16_t* ldst1 = (half ? &lsV[1][0] : &lsK[1][0]) + cr * 32 + ccol;

  async_b128(lds_off32(ldst0), gsrc);
  v8f acc = {};
  const int nk = KN / 32;
  for (int kc = 0; kc < nk; ++kc) {
    const int cur = kc & 1;
    wait_async0();
    __syncthreads();
    if (kc + 1 < nk)
      async_b128(lds_off32(cur ? ldst0 : ldst1), gsrc + (kc + 1) * 32);
    v16bf a = load_a(&lsK[cur][0] + d0 * 32, 32, lane);
    v16bf b = load_bT(&lsV[cur][0] + e0 * 32, 32, lane);
    acc = wmma_bf16(a, b, acc);
  }

  const int e = e0 + (lane & 15);
#pragma unroll
  for (int i = 0; i < 8; ++i) {
    const int d = d0 + i + ((lane >> 4) << 3);
    kvT[((size_t)bh * KD + e) * KD + d] = (bf16_t)acc[i];
  }
}

// ---------------------------------------------------------------------------
// k_sum[bh, d] = sum_n k[n, d]  (one wave per row, shuffle reduction)
// ---------------------------------------------------------------------------
__global__ __launch_bounds__(256)
void ksum_kernel(const bf16_t* __restrict__ kT, float* __restrict__ ksum) {
  const int gw   = blockIdx.x * 8 + (threadIdx.x >> 5);  // = bh*64 + d
  const int lane = threadIdx.x & 31;
  const bf16_t* row = kT + (size_t)gw * KN;
  float s = 0.f;
  for (int n = lane * 8; n < KN; n += 32 * 8) {
    v8bf v = *(const v8bf*)(row + n);
#pragma unroll
    for (int j = 0; j < 8; ++j) s += (float)v[j];
  }
#pragma unroll
  for (int o = 16; o > 0; o >>= 1) s += __shfl_xor(s, o, 32);
  if (lane == 0) ksum[gw] = s;
}

// ---------------------------------------------------------------------------
// out = (q @ kv) / (q . k_sum + eps), scattered to attn[B,N,C] in bf16.
// ---------------------------------------------------------------------------
__global__ __launch_bounds__(128)
void attn_kernel(const bf16_t* __restrict__ q_ws,   // [BH, N, D]
                 const bf16_t* __restrict__ kvT,    // [BH, E, D]
                 const float*  __restrict__ ksum,   // [BH, D]
                 bf16_t* __restrict__ attn) {       // [B*N, C]
  const int bh   = blockIdx.x;
  const int nt   = blockIdx.y;            // 16-token tile
  const int lane = threadIdx.x & 31;
  const int wave = threadIdx.x >> 5;      // e-tile 0..3
  const int b    = bh / KH;
  const int h    = bh % KH;

  const bf16_t* qbase  = q_ws + ((size_t)bh * KN + nt * 16) * KD;   // ld=64
  const bf16_t* kvbase = kvT + ((size_t)bh * KD + wave * 16) * KD;  // ld=64

  v8f acc = {};
#pragma unroll
  for (int k = 0; k < KD; k += 32) {
    v16bf a  = load_a(qbase + k, KD, lane);
    v16bf bm = load_bT(kvbase + k, KD, lane);
    acc = wmma_bf16(a, bm, acc);
  }

  // normalizer[token] = sum_d q[token,d]*ksum[d]; lanes (t, t+16) split D.
  const int tok = lane & 15;
  const bf16_t* qrow = qbase + tok * KD + ((lane >> 4) * 32);
  const float*  ks   = ksum + (size_t)bh * KD + ((lane >> 4) * 32);
  float nrm = 0.f;
#pragma unroll
  for (int j = 0; j < 32; ++j) nrm += (float)qrow[j] * ks[j];
  nrm += __shfl_xor(nrm, 16, 32);

  const int e = wave * 16 + (lane & 15);
#pragma unroll
  for (int i = 0; i < 8; ++i) {
    const int m  = i + ((lane >> 4) << 3);       // token in tile
    const float nm = __shfl(nrm, m, 32);         // lane m holds token m's norm
    const float val = acc[i] / (nm + 1e-8f);
    const int ntok = nt * 16 + m;
    attn[((size_t)b * KN + ntok) * KC + h * KD + e] = (bf16_t)val;
  }
}

// ---------------------------------------------------------------------------
// Projection GEMM: y = attn @ W_proj^T + b_proj, fp32 output.
// ---------------------------------------------------------------------------
__global__ __launch_bounds__(256)
void proj_gemm_kernel(const bf16_t* __restrict__ A,    // [M, C]
                      const bf16_t* __restrict__ W,    // [C, C] (B^T layout)
                      const float*  __restrict__ bias, // [C]
                      float* __restrict__ out) {       // [M, C]
  __shared__ bf16_t lsA[2][64 * 32];
  __shared__ bf16_t lsB[2][128 * 32];
  const int tid  = threadIdx.x;
  const int lane = tid & 31;
  const int wave = tid >> 5;
  const int wm   = wave >> 2, wn = wave & 3;
  const int m0   = blockIdx.y * 64;
  const int n0   = blockIdx.x * 128;

  v8f acc[2][2] = {};
  gemm_pipeline(A, W, KC, &lsA[0][0], &lsB[0][0], m0, n0, tid, lane, wm, wn, acc);

#pragma unroll
  for (int mi = 0; mi < 2; ++mi)
#pragma unroll
    for (int ni = 0; ni < 2; ++ni) {
      v8f a = acc[mi][ni];
      const int col = n0 + wn * 32 + ni * 16 + (lane & 15);
      const float bs = bias[col];
#pragma unroll
      for (int i = 0; i < 8; ++i) {
        const int m = m0 + wm * 32 + mi * 16 + i + ((lane >> 4) << 3);
        out[(size_t)m * KC + col] = a[i] + bs;
      }
    }
}

// ---------------------------------------------------------------------------
extern "C" void kernel_launch(void* const* d_in, const int* in_sizes, int n_in,
                              void* d_out, int out_size, void* d_ws, size_t ws_size,
                              hipStream_t stream) {
  const float* x           = (const float*)d_in[0];  // [B,N,C]
  const float* W_qkv       = (const float*)d_in[1];  // [3C,C]
  const float* W_proj      = (const float*)d_in[2];  // [C,C]
  const float* b_proj      = (const float*)d_in[3];  // [C]
  const float* temperature = (const float*)d_in[4];  // [H]

  char* ws = (char*)d_ws;
  bf16_t* xbf      = (bf16_t*)(ws + OFF_XBF);
  bf16_t* wqkv_bf  = (bf16_t*)(ws + OFF_WQKV);
  bf16_t* wproj_bf = (bf16_t*)(ws + OFF_WPROJ);
  bf16_t* q_ws     = (bf16_t*)(ws + OFF_QWS);
  bf16_t* kT_ws    = (bf16_t*)(ws + OFF_KT);
  bf16_t* vT_ws    = (bf16_t*)(ws + OFF_VT);
  bf16_t* kv_ws    = (bf16_t*)(ws + OFF_KV);
  float*  ksum     = (float*)(ws + OFF_KSUM);
  bf16_t* attn_ws  = (bf16_t*)(ws + OFF_ATTN);   // aliases xbf (dead by then)

  // 1) fp32 -> bf16 converts
  {
    int n = KM * KC;
    f32_to_bf16_kernel<<<dim3(n / (256 * 4)), dim3(256), 0, stream>>>(x, xbf, n);
    n = KCQ * KC;
    f32_to_bf16_kernel<<<dim3(n / (256 * 4)), dim3(256), 0, stream>>>(W_qkv, wqkv_bf, n);
    n = KC * KC;
    f32_to_bf16_kernel<<<dim3(n / (256 * 4)), dim3(256), 0, stream>>>(W_proj, wproj_bf, n);
  }

  // 2) QKV projection + ReLU(q/t), ReLU(k/t), transposed k/v stores
  qkv_gemm_kernel<<<dim3(KCQ / 128, KM / 64), dim3(256), 0, stream>>>(
      xbf, wqkv_bf, temperature, q_ws, kT_ws, vT_ws);

  // 3) kv state and k_sum
  kv_kernel<<<dim3(KB * KH), dim3(512), 0, stream>>>(kT_ws, vT_ws, kv_ws);
  ksum_kernel<<<dim3(KB * KH * KD / 8), dim3(256), 0, stream>>>(kT_ws, ksum);

  // 4) out = (q @ kv) / (q . ksum + eps)
  attn_kernel<<<dim3(KB * KH, KN / 16), dim3(128), 0, stream>>>(
      q_ws, kv_ws, ksum, attn_ws);

  // 5) projection + bias -> fp32 output
  proj_gemm_kernel<<<dim3(KC / 128, KM / 64), dim3(256), 0, stream>>>(
      attn_ws, wproj_bf, b_proj, (float*)d_out);
}
